// MultiHeadAttention_33904471835160
// MI455X (gfx1250) — matrix operand order repaired and checked by reading, not compiled, not run on hardware
//
#include <hip/hip_runtime.h>
#include <hip/hip_bf16.h>
#include <stdint.h>

#define HD    16
#define DM    1024
#define DKV   64
#define SEQ   2048
#define BATCH 2
#define MTOT  (BATCH * SEQ)      // 4096 rows
#define INV_TEMP 0.35355339059327373f  // 1 / 64^0.25

typedef __attribute__((ext_vector_type(16))) __bf16 v16bf;
typedef __attribute__((ext_vector_type(8)))  float  v8f;

union V16 {
    v16bf v;
    uint4 u[2];
};

__device__ __forceinline__ unsigned short f2bf(float f) {
    unsigned int u = __float_as_uint(f);
    u += 0x7FFFu + ((u >> 16) & 1u);          // round-to-nearest-even
    return (unsigned short)(u >> 16);
}

// ---------------------------------------------------------------- fp32 -> bf16
__global__ void cvt_bf16_kernel(const float* __restrict__ in,
                                unsigned short* __restrict__ out, int n) {
    int i = blockIdx.x * blockDim.x + threadIdx.x;
    if (i < n) out[i] = f2bf(in[i]);
}

// -------------------------------------------------- LayerNorm(q) -> bf16 rows
__global__ void layernorm_kernel(const float* __restrict__ q,
                                 const float* __restrict__ gamma,
                                 const float* __restrict__ beta,
                                 unsigned short* __restrict__ out) {
    __shared__ float red[256];
    int row = blockIdx.x;
    const float* r = q + (size_t)row * DM;
    float x[4];
    float s = 0.f;
#pragma unroll
    for (int i = 0; i < 4; i++) { x[i] = r[threadIdx.x + i * 256]; s += x[i]; }
    red[threadIdx.x] = s; __syncthreads();
    for (int off = 128; off > 0; off >>= 1) {
        if (threadIdx.x < off) red[threadIdx.x] += red[threadIdx.x + off];
        __syncthreads();
    }
    float mu = red[0] * (1.0f / DM);
    __syncthreads();
    float vv = 0.f;
#pragma unroll
    for (int i = 0; i < 4; i++) { float d = x[i] - mu; vv += d * d; }
    red[threadIdx.x] = vv; __syncthreads();
    for (int off = 128; off > 0; off >>= 1) {
        if (threadIdx.x < off) red[threadIdx.x] += red[threadIdx.x + off];
        __syncthreads();
    }
    float rs = rsqrtf(red[0] * (1.0f / DM) + 1e-6f);
#pragma unroll
    for (int i = 0; i < 4; i++) {
        int c = threadIdx.x + i * 256;
        out[(size_t)row * DM + c] = f2bf((x[i] - mu) * rs * gamma[c] + beta[c]);
    }
}

// -------------------------------------- W (K x N fp32) -> WT (N x K bf16)
__global__ void transpose_cvt_kernel(const float* __restrict__ W,
                                     unsigned short* __restrict__ WT) {
    int i = blockIdx.x * blockDim.x + threadIdx.x;   // over DM*DM
    int kk = i >> 10, n = i & 1023;
    WT[(size_t)n * DM + kk] = f2bf(W[i]);
}

// ---- vproj (MTOT x DM bf16, col = h*64+d) -> vhT[(b*16+h)*64 + d][SEQ] bf16
__global__ void vtrans_kernel(const unsigned short* __restrict__ vproj,
                              unsigned short* __restrict__ vhT) {
    int i = blockIdx.x * blockDim.x + threadIdx.x;   // BATCH*HD*DKV*SEQ
    int l  = i & (SEQ - 1);
    int t  = i >> 11;
    int d  = t & 63;
    int bh = t >> 6;
    int b = bh >> 4, h = bh & 15;
    vhT[i] = vproj[((size_t)(b * SEQ + l)) * DM + h * DKV + d];
}

// ---------------- C_bf16 (MTOT x DM) = scale * A (MTOT x DM) @ BT(N x K)^T
__global__ void gemm_bf16_kernel(const unsigned short* __restrict__ A,
                                 const unsigned short* __restrict__ BT,
                                 unsigned short* __restrict__ C, float scale) {
    int m0 = blockIdx.x * 16;
    int n0 = blockIdx.y * 64;
    int lane = threadIdx.x, hl = lane & 15, hi = lane >> 4;
    v8f acc[4] = {};
    const unsigned short* arow = A + (size_t)(m0 + hl) * DM;
    for (int k0 = 0; k0 < DM; k0 += 32) {
        __builtin_prefetch(arow + k0 + 128, 0, 1);
        V16 a;
        a.u[0] = *(const uint4*)(arow + k0 + hi * 8);
        a.u[1] = *(const uint4*)(arow + k0 + 16 + hi * 8);
#pragma unroll
        for (int nt = 0; nt < 4; nt++) {
            V16 b;
            const unsigned short* brow =
                BT + (size_t)(n0 + nt * 16 + hl) * DM + k0 + hi * 8;
            b.u[0] = *(const uint4*)brow;
            b.u[1] = *(const uint4*)(brow + 16);
            acc[nt] = __builtin_amdgcn_wmma_f32_16x16x32_bf16(
                false, a.v, false, b.v, (short)0, acc[nt], false, false);
        }
    }
#pragma unroll
    for (int nt = 0; nt < 4; nt++)
#pragma unroll
        for (int j = 0; j < 8; j++)
            C[(size_t)(m0 + j + 8 * hi) * DM + n0 + nt * 16 + hl] =
                f2bf(acc[nt][j] * scale);
}

// ------------- out (fp32) = A_bf16 @ BT^T + resid   (final FC + residual)
__global__ void gemm_out_kernel(const unsigned short* __restrict__ A,
                                const unsigned short* __restrict__ BT,
                                const float* __restrict__ resid,
                                float* __restrict__ out) {
    int m0 = blockIdx.x * 16;
    int n0 = blockIdx.y * 64;
    int lane = threadIdx.x, hl = lane & 15, hi = lane >> 4;
    v8f acc[4] = {};
    const unsigned short* arow = A + (size_t)(m0 + hl) * DM;
    for (int k0 = 0; k0 < DM; k0 += 32) {
        V16 a;
        a.u[0] = *(const uint4*)(arow + k0 + hi * 8);
        a.u[1] = *(const uint4*)(arow + k0 + 16 + hi * 8);
#pragma unroll
        for (int nt = 0; nt < 4; nt++) {
            V16 b;
            const unsigned short* brow =
                BT + (size_t)(n0 + nt * 16 + hl) * DM + k0 + hi * 8;
            b.u[0] = *(const uint4*)brow;
            b.u[1] = *(const uint4*)(brow + 16);
            acc[nt] = __builtin_amdgcn_wmma_f32_16x16x32_bf16(
                false, a.v, false, b.v, (short)0, acc[nt], false, false);
        }
    }
#pragma unroll
    for (int nt = 0; nt < 4; nt++)
#pragma unroll
        for (int j = 0; j < 8; j++) {
            size_t idx = (size_t)(m0 + j + 8 * hi) * DM + n0 + nt * 16 + hl;
            out[idx] = acc[nt][j] + resid[idx];
        }
}

// ---- fused: S = (Q/T) Kᵀ ; mask ; store attn(fp32) ; O += S @ V ; store O_bf16
__global__ void attn_kernel(const unsigned short* __restrict__ qproj,
                            const unsigned short* __restrict__ kproj,
                            const unsigned short* __restrict__ vhT,
                            const int* __restrict__ mask,
                            float* __restrict__ attn_out,
                            unsigned short* __restrict__ headout) {
    __shared__ __align__(16) unsigned short sS[16 * 32];
    int q0 = blockIdx.x * 16;
    int h  = blockIdx.y;
    int b  = blockIdx.z;
    int lane = threadIdx.x, hl = lane & 15, hi = lane >> 4;

    V16 aq[2];
    {
        const unsigned short* qrow =
            qproj + ((size_t)(b * SEQ + q0 + hl)) * DM + h * DKV;
        aq[0].u[0] = *(const uint4*)(qrow + hi * 8);
        aq[0].u[1] = *(const uint4*)(qrow + 16 + hi * 8);
        aq[1].u[0] = *(const uint4*)(qrow + 32 + hi * 8);
        aq[1].u[1] = *(const uint4*)(qrow + 48 + hi * 8);
    }
    v8f o[4] = {};
    const size_t attn_row0 = (size_t)(b * HD + h) * SEQ;  // row units

    for (int kc0 = 0; kc0 < SEQ; kc0 += 32) {
        v8f s[2] = {};
#pragma unroll
        for (int half = 0; half < 2; half++) {
#pragma unroll
            for (int dd = 0; dd < 2; dd++) {
                V16 bk;
                const unsigned short* krow =
                    kproj + ((size_t)(b * SEQ + kc0 + half * 16 + hl)) * DM +
                    h * DKV + dd * 32 + hi * 8;
                bk.u[0] = *(const uint4*)krow;
                bk.u[1] = *(const uint4*)(krow + 16);
                s[half] = __builtin_amdgcn_wmma_f32_16x16x32_bf16(
                    false, aq[dd].v, false, bk.v, (short)0, s[half], false, false);
            }
        }
        // mask, emit attn (fp32), stage S in LDS as bf16 row-major 16x32
#pragma unroll
        for (int half = 0; half < 2; half++) {
#pragma unroll
            for (int j = 0; j < 8; j++) {
                int r = j + 8 * hi;
                int c = half * 16 + hl;
                float val = s[half][j];
                int mv = mask[((size_t)b * SEQ + q0 + r) * SEQ + kc0 + c];
                if (mv == 0) val = -1e9f;
                attn_out[(attn_row0 + q0 + r) * SEQ + kc0 + c] = val;
                sS[r * 32 + c] = f2bf(val);
            }
        }
        __syncthreads();
        V16 as;
        as.u[0] = *(const uint4*)(&sS[hl * 32 + hi * 8]);
        as.u[1] = *(const uint4*)(&sS[hl * 32 + 16 + hi * 8]);
#pragma unroll
        for (int nt = 0; nt < 4; nt++) {
            V16 bv;
            const unsigned short* vrow =
                vhT + ((size_t)((b * HD + h) * DKV + nt * 16 + hl)) * SEQ +
                kc0 + hi * 8;
            bv.u[0] = *(const uint4*)vrow;
            bv.u[1] = *(const uint4*)(vrow + 16);
            o[nt] = __builtin_amdgcn_wmma_f32_16x16x32_bf16(
                false, as.v, false, bv.v, (short)0, o[nt], false, false);
        }
        __syncthreads();
    }
#pragma unroll
    for (int nt = 0; nt < 4; nt++)
#pragma unroll
        for (int j = 0; j < 8; j++)
            headout[((size_t)(b * SEQ + q0 + j + 8 * hi)) * DM +
                    h * DKV + nt * 16 + hl] = f2bf(o[nt][j]);
}

extern "C" void kernel_launch(void* const* d_in, const int* in_sizes, int n_in,
                              void* d_out, int out_size, void* d_ws, size_t ws_size,
                              hipStream_t stream) {
    (void)in_sizes; (void)n_in; (void)out_size; (void)ws_size;
    const float* q     = (const float*)d_in[0];
    const float* k     = (const float*)d_in[1];
    const float* v     = (const float*)d_in[2];
    const int*   mask  = (const int*)  d_in[3];
    const float* Wq    = (const float*)d_in[4];
    const float* Wk    = (const float*)d_in[5];
    const float* Wv    = (const float*)d_in[6];
    const float* Wfc   = (const float*)d_in[7];
    const float* gamma = (const float*)d_in[8];
    const float* beta  = (const float*)d_in[9];

    float* out      = (float*)d_out;
    float* attn_out = out + (size_t)MTOT * DM;

    char* ws = (char*)d_ws;
    const size_t SZ_ACT = (size_t)MTOT * DM * sizeof(unsigned short); // 8 MB
    const size_t SZ_W   = (size_t)DM * DM * sizeof(unsigned short);   // 2 MB
    unsigned short* qn_bf   = (unsigned short*)ws; ws += SZ_ACT;
    unsigned short* k_bf    = (unsigned short*)ws; ws += SZ_ACT;
    unsigned short* v_bf    = (unsigned short*)ws; ws += SZ_ACT;
    unsigned short* WqT     = (unsigned short*)ws; ws += SZ_W;
    unsigned short* WkT     = (unsigned short*)ws; ws += SZ_W;
    unsigned short* WvT     = (unsigned short*)ws; ws += SZ_W;
    unsigned short* WfcT    = (unsigned short*)ws; ws += SZ_W;
    unsigned short* qprojB  = (unsigned short*)ws; ws += SZ_ACT;
    unsigned short* kprojB  = (unsigned short*)ws; ws += SZ_ACT;
    unsigned short* vprojB  = (unsigned short*)ws; ws += SZ_ACT;
    unsigned short* vhT     = (unsigned short*)ws; ws += SZ_ACT;
    unsigned short* headout = (unsigned short*)ws; ws += SZ_ACT;

    const int nAct = MTOT * DM;                 // 4,194,304

    hipLaunchKernelGGL(layernorm_kernel, dim3(MTOT), dim3(256), 0, stream,
                       q, gamma, beta, qn_bf);
    hipLaunchKernelGGL(cvt_bf16_kernel, dim3(nAct / 256), dim3(256), 0, stream,
                       k, k_bf, nAct);
    hipLaunchKernelGGL(cvt_bf16_kernel, dim3(nAct / 256), dim3(256), 0, stream,
                       v, v_bf, nAct);
    hipLaunchKernelGGL(transpose_cvt_kernel, dim3(DM * DM / 256), dim3(256), 0, stream, Wq,  WqT);
    hipLaunchKernelGGL(transpose_cvt_kernel, dim3(DM * DM / 256), dim3(256), 0, stream, Wk,  WkT);
    hipLaunchKernelGGL(transpose_cvt_kernel, dim3(DM * DM / 256), dim3(256), 0, stream, Wv,  WvT);
    hipLaunchKernelGGL(transpose_cvt_kernel, dim3(DM * DM / 256), dim3(256), 0, stream, Wfc, WfcT);

    dim3 gemmGrid(MTOT / 16, DM / 64);
    hipLaunchKernelGGL(gemm_bf16_kernel, gemmGrid, dim3(32), 0, stream,
                       qn_bf, WqT, qprojB, INV_TEMP);   // fold 1/TEMP into Q
    hipLaunchKernelGGL(gemm_bf16_kernel, gemmGrid, dim3(32), 0, stream,
                       k_bf, WkT, kprojB, 1.0f);
    hipLaunchKernelGGL(gemm_bf16_kernel, gemmGrid, dim3(32), 0, stream,
                       v_bf, WvT, vprojB, 1.0f);

    hipLaunchKernelGGL(vtrans_kernel, dim3(nAct / 256), dim3(256), 0, stream,
                       vprojB, vhT);

    hipLaunchKernelGGL(attn_kernel, dim3(SEQ / 16, HD, BATCH), dim3(32), 0, stream,
                       qprojB, kprojB, vhT, mask, attn_out, headout);

    hipLaunchKernelGGL(gemm_out_kernel, gemmGrid, dim3(32), 0, stream,
                       headout, WfcT, q, out);
}
